// TopKWindowAttentionLayerV2_39917426049204
// MI455X (gfx1250) — matrix operand order repaired and checked by reading, not compiled, not run on hardware
//
#include <hip/hip_runtime.h>
#include <hip/hip_bf16.h>
#include <math.h>

// ---------------------------------------------------------------------------
// TopK window attention layer for MI455X (gfx1250, wave32, WMMA bf16).
// All GEMMs (QKV, sim, merge, fc1, fc2) and both attention contractions run
// through v_wmma_f32_16x16x32_bf16 with fp32 accumulation. GEMM uses a 64x64
// output tile per wave (16 WMMAs per K-step, 0.5KB of L2 traffic per WMMA).
// Attention gathers top-k key/value windows on the fly (no 170MB keys/vals
// materialization) with a two-pass softmax and an LDS P-tile transpose.
// ---------------------------------------------------------------------------

typedef __bf16 v16bf __attribute__((ext_vector_type(16)));
typedef float  v8f   __attribute__((ext_vector_type(8)));

#define DEV __device__ __forceinline__

constexpr int L_    = 12544;   // 112*112 tokens
constexpr int D_    = 256;     // d_model
constexpr int D2_   = 512;     // 2*d
constexpr int NW_   = 256;     // windows (16x16)
constexpr int WSZ_  = 49;      // tokens per window (7x7)
constexpr int NH_   = 8;       // heads
constexpr int DH_   = 32;      // head dim
constexpr int TK_   = 8;       // top-k
constexpr int SKEY_ = TK_ * WSZ_ + NW_;  // 648 keys per window
constexpr int NCH_  = (SKEY_ + 31) / 32; // 21 key chunks of 32

DEV __bf16 f2bf(float f) {
  unsigned u = __float_as_uint(f);
  unsigned r = (u + 0x7FFFu + ((u >> 16) & 1u)) >> 16;   // round-to-nearest-even
  unsigned short s = (unsigned short)r;
  return __builtin_bit_cast(__bf16, s);
}
DEV float bf2f(__bf16 h) {
  unsigned short s = __builtin_bit_cast(unsigned short, h);
  return __uint_as_float(((unsigned)s) << 16);
}
DEV v16bf bfzero16() {
  v16bf v;
#pragma unroll
  for (int i = 0; i < 16; ++i) v[i] = f2bf(0.0f);
  return v;
}
DEV v8f fzero8() {
  v8f v;
#pragma unroll
  for (int i = 0; i < 8; ++i) v[i] = 0.0f;
  return v;
}
DEV v8f wmma_bf(v16bf a, v16bf b, v8f c) {
  // D = A(16x32 bf16) * B(32x16 bf16) + C(16x16 f32)
  return __builtin_amdgcn_wmma_f32_16x16x32_bf16(false, a, false, b, (short)0, c,
                                                 false, false);
}
// monotonic float<->uint encoding for atomicMax-based row maxima
DEV unsigned fenc(float f) {
  unsigned u = __float_as_uint(f);
  return (u & 0x80000000u) ? ~u : (u | 0x80000000u);
}
DEV float fdec(unsigned e) {
  unsigned u = (e & 0x80000000u) ? (e & 0x7FFFFFFFu) : ~e;
  return __uint_as_float(u);
}

// ---------------------------------------------------------------------------
// Cast / transpose kernels
// ---------------------------------------------------------------------------
__global__ __launch_bounds__(256) void cast_xs_kernel(
    const float* __restrict__ x, const float* __restrict__ src,
    __bf16* __restrict__ xfb, __bf16* __restrict__ sfb, __bf16* __restrict__ cat) {
  int i = blockIdx.x * 256 + threadIdx.x;          // over L*D
  if (i >= L_ * D_) return;
  int l = i >> 8, c = i & 255;
  __bf16 h = f2bf(x[(size_t)c * L_ + l]);
  xfb[i] = h;
  cat[(size_t)l * D2_ + c] = h;                    // first half of concat
  sfb[i] = f2bf(src[(size_t)c * L_ + l]);
}

__global__ __launch_bounds__(256) void cast_w_kernel(
    const float* __restrict__ src, __bf16* __restrict__ dst, int n) {
  int i = blockIdx.x * 256 + threadIdx.x;
  if (i < n) dst[i] = f2bf(src[i]);
}

// ---------------------------------------------------------------------------
// WMMA GEMM, 64x64 tile per wave: C[M,N] = A[M,K] * W[N,K]^T (bf16 in).
// 4 A-frags x 4 B-frags -> 16 WMMAs per 32-deep K step.
// ---------------------------------------------------------------------------
enum { EP_BF16 = 0, EP_F32 = 1, EP_WIN_BF16 = 2, EP_BIAS_RELU_T = 3, EP_BIAS_F32 = 4 };

template <int EP>
__global__ __launch_bounds__(32) void gemm_wmma_kernel(
    const __bf16* __restrict__ A, const __bf16* __restrict__ W,
    const float* __restrict__ bias, float* __restrict__ Cf,
    __bf16* __restrict__ Cb, int M, int N, int K) {
  const int nt = blockIdx.x, mt = blockIdx.y;      // 64x64 tile coords
  const int lane = threadIdx.x, half = lane >> 4, l16 = lane & 15;

  v8f acc[4][4];
#pragma unroll
  for (int i = 0; i < 4; ++i)
#pragma unroll
    for (int j = 0; j < 4; ++j) acc[i][j] = fzero8();

  for (int k0 = 0; k0 < K; k0 += 32) {
    v16bf afr[4], bfr[4];
#pragma unroll
    for (int i = 0; i < 4; ++i) {
      afr[i] = bfzero16();
      int arow = mt * 64 + i * 16 + l16;
      if (arow < M) {
        const __bf16* ap = A + (size_t)arow * K + k0;
        if (k0 + 32 < K) __builtin_prefetch(ap + 32, 0, 1);
#pragma unroll
        for (int p = 0; p < 8; ++p) {
          int c = (p < 4 ? 2 * p : 16 + 2 * (p - 4)) + half * 8;  // A 16x32 layout
          afr[i][2 * p] = ap[c]; afr[i][2 * p + 1] = ap[c + 1];
        }
      }
    }
#pragma unroll
    for (int j = 0; j < 4; ++j) {
      bfr[j] = bfzero16();
      int bcol = nt * 64 + j * 16 + l16;
      if (bcol < N) {
        const __bf16* bp = W + (size_t)bcol * K + k0;
        if (k0 + 32 < K) __builtin_prefetch(bp + 32, 0, 1);
#pragma unroll
        for (int p = 0; p < 8; ++p) {
          int c = (p < 4 ? 2 * p : 16 + 2 * (p - 4)) + half * 8;  // B 32x16 layout
          bfr[j][2 * p] = bp[c]; bfr[j][2 * p + 1] = bp[c + 1];
        }
      }
    }
#pragma unroll
    for (int i = 0; i < 4; ++i)
#pragma unroll
      for (int j = 0; j < 4; ++j) acc[i][j] = wmma_bf(afr[i], bfr[j], acc[i][j]);
  }

#pragma unroll
  for (int i = 0; i < 4; ++i) {
#pragma unroll
    for (int j = 0; j < 4; ++j) {
      const int col = nt * 64 + j * 16 + l16;
#pragma unroll
      for (int r = 0; r < 8; ++r) {
        int row = mt * 64 + i * 16 + r + half * 8;                 // C layout
        if (row >= M || col >= N) continue;
        float v = acc[i][j][r];
        if constexpr (EP == EP_BF16) {
          Cb[(size_t)row * N + col] = f2bf(v);
        } else if constexpr (EP == EP_F32) {
          Cf[(size_t)row * N + col] = v;
        } else if constexpr (EP == EP_WIN_BF16) {
          // spatial row-major l -> window-major l'
          int y = row / 112, xx = row % 112;
          int lw = ((y / 7) * 16 + (xx / 7)) * WSZ_ + (y % 7) * 7 + (xx % 7);
          Cb[(size_t)lw * N + col] = f2bf(v);
        } else if constexpr (EP == EP_BIAS_RELU_T) {
          float t = v + bias[col];
          Cf[(size_t)col * M + row] = t > 0.0f ? t : 0.0f;         // NCHW for dwconv
        } else if constexpr (EP == EP_BIAS_F32) {
          Cf[(size_t)row * N + col] = v + bias[col];
        }
      }
    }
  }
}

// ---------------------------------------------------------------------------
// Window means (q/k/v, windowed layout) -> bf16
// ---------------------------------------------------------------------------
__global__ __launch_bounds__(256) void means_kernel(
    const __bf16* __restrict__ qw, const __bf16* __restrict__ kw,
    const __bf16* __restrict__ vw, __bf16* __restrict__ qmb,
    __bf16* __restrict__ kmb, __bf16* __restrict__ vmb) {
  int w = blockIdx.x, c = threadIdx.x;
  float sq = 0.f, sk = 0.f, sv = 0.f;
  for (int t = 0; t < WSZ_; ++t) {
    size_t o = ((size_t)(w * WSZ_ + t)) * D_ + c;
    sq += bf2f(qw[o]); sk += bf2f(kw[o]); sv += bf2f(vw[o]);
  }
  const float inv = 1.0f / 49.0f;
  qmb[(size_t)w * D_ + c] = f2bf(sq * inv);
  kmb[(size_t)w * D_ + c] = f2bf(sk * inv);
  vmb[(size_t)w * D_ + c] = f2bf(sv * inv);
}

// ---------------------------------------------------------------------------
// Top-8 per row of sim (matches lax.top_k ordering, ties -> lowest index)
// ---------------------------------------------------------------------------
__global__ __launch_bounds__(32) void topk_kernel(const float* __restrict__ sim,
                                                  int* __restrict__ idx) {
  int row = blockIdx.x, lane = threadIdx.x;
  float v[8];
#pragma unroll
  for (int i = 0; i < 8; ++i) v[i] = sim[(size_t)row * NW_ + i * 32 + lane];
  for (int t = 0; t < TK_; ++t) {
    float bv = -3.0e38f; int bj = 0x7fffffff;
#pragma unroll
    for (int i = 0; i < 8; ++i) {
      int j = i * 32 + lane;
      if (v[i] > bv || (v[i] == bv && j < bj)) { bv = v[i]; bj = j; }
    }
    for (int off = 16; off > 0; off >>= 1) {
      float ov = __shfl_xor(bv, off, 32);
      int   oj = __shfl_xor(bj, off, 32);
      if (ov > bv || (ov == bv && oj < bj)) { bv = ov; bj = oj; }
    }
    if (lane == 0) idx[row * TK_ + t] = bj;
    if ((bj & 31) == lane) v[bj >> 5] = -3.0e38f;
  }
}

// ---------------------------------------------------------------------------
// Attention: one wave per (window, head). Two-pass softmax, on-the-fly gather.
// ---------------------------------------------------------------------------
__global__ __launch_bounds__(32) void attn_kernel(
    const __bf16* __restrict__ qw, const __bf16* __restrict__ kw,
    const __bf16* __restrict__ vw, const __bf16* __restrict__ kmb,
    const __bf16* __restrict__ vmb, const int* __restrict__ idx,
    __bf16* __restrict__ msg) {
  const int w = blockIdx.x, h = blockIdx.y;
  const int lane = threadIdx.x, half = lane >> 4, l16 = lane & 15;

  __shared__ int widx[TK_];
  __shared__ unsigned rmu[64];
  __shared__ float rmax[64];
  __shared__ float rsum[64];
  __shared__ __bf16 ptile[16 * 32];

  if (lane < TK_) widx[lane] = idx[w * TK_ + lane];
  rmu[lane] = 0u;  rmu[lane + 32] = 0u;
  rsum[lane] = 0.f; rsum[lane + 32] = 0.f;
  __syncthreads();

  // Q fragments: dh=32 == one full WMMA K step.
  v16bf qf[4];
#pragma unroll
  for (int t = 0; t < 4; ++t) {
    qf[t] = bfzero16();
    int tok = t * 16 + l16;
    if (tok < WSZ_) {
      const __bf16* qp = qw + ((size_t)(w * WSZ_ + tok)) * D_ + h * DH_;
#pragma unroll
      for (int p = 0; p < 8; ++p) {
        int c = (p < 4 ? 2 * p : 16 + 2 * (p - 4)) + half * 8;
        qf[t][2 * p] = qp[c]; qf[t][2 * p + 1] = qp[c + 1];
      }
    }
  }

  auto kbase = [&](int j) -> const __bf16* {
    if (j < TK_ * WSZ_) {
      int wn = widx[j / WSZ_]; int tk = j % WSZ_;
      return kw + ((size_t)(wn * WSZ_ + tk)) * D_ + h * DH_;
    }
    return kmb + (size_t)(j - TK_ * WSZ_) * D_ + h * DH_;
  };
  auto vbase = [&](int j) -> const __bf16* {
    if (j < TK_ * WSZ_) {
      int wn = widx[j / WSZ_]; int tk = j % WSZ_;
      return vw + ((size_t)(wn * WSZ_ + tk)) * D_ + h * DH_;
    }
    return vmb + (size_t)(j - TK_ * WSZ_) * D_ + h * DH_;
  };
  auto load_kfrag = [&](int jb) -> v16bf {   // B = K^T (32 dims x 16 keys)
    v16bf b = bfzero16();
    int j = jb + l16;
    if (j < SKEY_) {
      const __bf16* kp = kbase(j);
#pragma unroll
      for (int p = 0; p < 8; ++p) {
        int c = (p < 4 ? 2 * p : 16 + 2 * (p - 4)) + half * 8;
        b[2 * p] = kp[c]; b[2 * p + 1] = kp[c + 1];
      }
    }
    return b;
  };
  auto load_vfrag = [&](int kc, int nt) -> v16bf {  // B = V (32 keys x 16 dims)
    v16bf b = bfzero16();
    int n = nt * 16 + l16;
#pragma unroll
    for (int p = 0; p < 8; ++p) {
#pragma unroll
      for (int e = 0; e < 2; ++e) {
        int kk = (p < 4 ? 2 * p : 16 + 2 * (p - 4)) + half * 8 + e;
        int j = kc * 32 + kk;
        if (j < SKEY_) b[2 * p + e] = vbase(j)[n];
      }
    }
    return b;
  };

  const float scale = 0.17677669529663687f;  // 1/sqrt(32)

  // Pass 1: row maxima.
  for (int kc = 0; kc < NCH_; ++kc) {
    v16bf b0 = load_kfrag(kc * 32);
    v16bf b1 = load_kfrag(kc * 32 + 16);
#pragma unroll
    for (int t = 0; t < 4; ++t) {
      v8f s0 = wmma_bf(qf[t], b0, fzero8());
      v8f s1 = wmma_bf(qf[t], b1, fzero8());
      int j0 = kc * 32 + l16, j1 = j0 + 16;
#pragma unroll
      for (int r = 0; r < 8; ++r) {
        int row = t * 16 + r + half * 8;
        float m = -3.0e38f;
        if (j0 < SKEY_) m = fmaxf(m, s0[r] * scale);
        if (j1 < SKEY_) m = fmaxf(m, s1[r] * scale);
        atomicMax(&rmu[row], fenc(m));
      }
    }
  }
  __syncthreads();
  rmax[lane] = fdec(rmu[lane]);
  rmax[lane + 32] = fdec(rmu[lane + 32]);
  __syncthreads();

  v8f acc[4][2];
#pragma unroll
  for (int t = 0; t < 4; ++t) { acc[t][0] = fzero8(); acc[t][1] = fzero8(); }

  // Pass 2: unnormalized probabilities + O accumulation (divide at the end).
  for (int kc = 0; kc < NCH_; ++kc) {
    v16bf b0 = load_kfrag(kc * 32);
    v16bf b1 = load_kfrag(kc * 32 + 16);
    v16bf vf0 = load_vfrag(kc, 0);
    v16bf vf1 = load_vfrag(kc, 1);
#pragma unroll
    for (int t = 0; t < 4; ++t) {
      v8f s0 = wmma_bf(qf[t], b0, fzero8());
      v8f s1 = wmma_bf(qf[t], b1, fzero8());
      int j0 = kc * 32 + l16, j1 = j0 + 16;
      __syncthreads();
#pragma unroll
      for (int r = 0; r < 8; ++r) {
        int m = r + half * 8;
        int row = t * 16 + m;
        float e0 = 0.f, e1 = 0.f;
        if (row < WSZ_) {
          if (j0 < SKEY_) e0 = __expf(s0[r] * scale - rmax[row]);
          if (j1 < SKEY_) e1 = __expf(s1[r] * scale - rmax[row]);
        }
        if (e0 != 0.f || e1 != 0.f) atomicAdd(&rsum[row], e0 + e1);
        ptile[m * 32 + l16] = f2bf(e0);
        ptile[m * 32 + 16 + l16] = f2bf(e1);
      }
      __syncthreads();
      v16bf pa;  // P tile re-read in A-fragment layout (LDS transpose)
#pragma unroll
      for (int p = 0; p < 8; ++p) {
        int c = (p < 4 ? 2 * p : 16 + 2 * (p - 4)) + half * 8;
        pa[2 * p] = ptile[l16 * 32 + c];
        pa[2 * p + 1] = ptile[l16 * 32 + c + 1];
      }
      acc[t][0] = wmma_bf(pa, vf0, acc[t][0]);
      acc[t][1] = wmma_bf(pa, vf1, acc[t][1]);
    }
  }
  __syncthreads();
#pragma unroll
  for (int t = 0; t < 4; ++t) {
#pragma unroll
    for (int nt = 0; nt < 2; ++nt) {
#pragma unroll
      for (int r = 0; r < 8; ++r) {
        int tok = t * 16 + r + half * 8;
        if (tok < WSZ_) {
          float val = acc[t][nt][r] / rsum[tok];
          msg[((size_t)(w * WSZ_ + tok)) * D_ + h * DH_ + nt * 16 + l16] = f2bf(val);
        }
      }
    }
  }
}

// ---------------------------------------------------------------------------
// LayerNorms
// ---------------------------------------------------------------------------
__global__ __launch_bounds__(256) void ln1_kernel(
    const float* __restrict__ merged, const float* __restrict__ g,
    const float* __restrict__ bb, __bf16* __restrict__ cat) {
  int l = blockIdx.x, c = threadIdx.x;
  __shared__ float red[256];
  float v = merged[(size_t)l * D_ + c];
  red[c] = v; __syncthreads();
  for (int s = 128; s > 0; s >>= 1) { if (c < s) red[c] += red[c + s]; __syncthreads(); }
  float mu = red[0] * (1.0f / 256.0f);
  __syncthreads();
  float d = v - mu;
  red[c] = d * d; __syncthreads();
  for (int s = 128; s > 0; s >>= 1) { if (c < s) red[c] += red[c + s]; __syncthreads(); }
  float inv = rsqrtf(red[0] * (1.0f / 256.0f) + 1e-5f);
  cat[(size_t)l * D2_ + D_ + c] = f2bf(d * inv * g[c] + bb[c]);
}

__global__ __launch_bounds__(256) void ln2_kernel(
    const float* __restrict__ t, const float* __restrict__ x,
    const float* __restrict__ g, const float* __restrict__ bb,
    float* __restrict__ out) {
  int l = blockIdx.x, c = threadIdx.x;
  __shared__ float red[256];
  float v = t[(size_t)l * D_ + c];
  red[c] = v; __syncthreads();
  for (int s = 128; s > 0; s >>= 1) { if (c < s) red[c] += red[c + s]; __syncthreads(); }
  float mu = red[0] * (1.0f / 256.0f);
  __syncthreads();
  float d = v - mu;
  red[c] = d * d; __syncthreads();
  for (int s = 128; s > 0; s >>= 1) { if (c < s) red[c] += red[c + s]; __syncthreads(); }
  float inv = rsqrtf(red[0] * (1.0f / 256.0f) + 1e-5f);
  float o = d * inv * g[c] + bb[c] + x[(size_t)c * L_ + l];
  out[(size_t)c * L_ + l] = o;                     // back to (b, d, h, w)
}

// ---------------------------------------------------------------------------
// Depthwise 3x3 'SAME' conv + exact GELU; NCHW in, [L, 512] bf16 out
// ---------------------------------------------------------------------------
__global__ __launch_bounds__(256) void dwconv_kernel(
    const float* __restrict__ tchw, const float* __restrict__ dww,
    const float* __restrict__ dwb, __bf16* __restrict__ t2) {
  int c = blockIdx.z;
  int x0 = blockIdx.x * 16, y0 = blockIdx.y * 16;
  int tx = threadIdx.x, ty = threadIdx.y;
  __shared__ float tile[18][18];
  for (int i = ty * 16 + tx; i < 18 * 18; i += 256) {
    int yy = y0 + i / 18 - 1, xx = x0 + i % 18 - 1;
    tile[i / 18][i % 18] =
        (yy >= 0 && yy < 112 && xx >= 0 && xx < 112)
            ? tchw[(size_t)c * L_ + yy * 112 + xx] : 0.0f;
  }
  __syncthreads();
  float a = dwb[c];
#pragma unroll
  for (int ky = 0; ky < 3; ++ky)
#pragma unroll
    for (int kx = 0; kx < 3; ++kx)
      a += dww[c * 9 + ky * 3 + kx] * tile[ty + ky][tx + kx];
  float gelu = 0.5f * a * (1.0f + erff(a * 0.70710678118654752f));
  int l = (y0 + ty) * 112 + (x0 + tx);
  t2[(size_t)l * D2_ + c] = f2bf(gelu);
}

// ---------------------------------------------------------------------------
// Host launch
// ---------------------------------------------------------------------------
extern "C" void kernel_launch(void* const* d_in, const int* in_sizes, int n_in,
                              void* d_out, int out_size, void* d_ws, size_t ws_size,
                              hipStream_t stream) {
  (void)in_sizes; (void)n_in; (void)out_size; (void)ws_size;
  const float* x     = (const float*)d_in[0];
  const float* src   = (const float*)d_in[1];
  const float* Wq    = (const float*)d_in[2];
  const float* Wk    = (const float*)d_in[3];
  const float* Wv    = (const float*)d_in[4];
  const float* Wm    = (const float*)d_in[5];
  const float* fc1w  = (const float*)d_in[6];
  const float* fc1b  = (const float*)d_in[7];
  const float* dww   = (const float*)d_in[8];
  const float* dwb   = (const float*)d_in[9];
  const float* fc2w  = (const float*)d_in[10];
  const float* fc2b  = (const float*)d_in[11];
  const float* n1g   = (const float*)d_in[12];
  const float* n1b   = (const float*)d_in[13];
  const float* n2g   = (const float*)d_in[14];
  const float* n2b   = (const float*)d_in[15];
  float* out = (float*)d_out;

  size_t off = 0;
  char* base = (char*)d_ws;
  auto alloc = [&](size_t bytes) -> char* {
    char* p = base + off;
    off += (bytes + 255) & ~(size_t)255;
    return p;
  };
  const size_t LD2 = (size_t)L_ * D_ * sizeof(__bf16);
  __bf16* xfb   = (__bf16*)alloc(LD2);
  __bf16* sfb   = (__bf16*)alloc(LD2);
  __bf16* wqb   = (__bf16*)alloc((size_t)D_ * D_ * 2);
  __bf16* wkb   = (__bf16*)alloc((size_t)D_ * D_ * 2);
  __bf16* wvb   = (__bf16*)alloc((size_t)D_ * D_ * 2);
  __bf16* wmb_  = (__bf16*)alloc((size_t)D_ * D_ * 2);
  __bf16* f1wb  = (__bf16*)alloc((size_t)D2_ * D2_ * 2);
  __bf16* f2wb  = (__bf16*)alloc((size_t)D_ * D2_ * 2);
  __bf16* qwin  = (__bf16*)alloc(LD2);
  __bf16* kwin  = (__bf16*)alloc(LD2);
  __bf16* vwin  = (__bf16*)alloc(LD2);
  __bf16* qmb   = (__bf16*)alloc((size_t)NW_ * D_ * 2);
  __bf16* kmb   = (__bf16*)alloc((size_t)NW_ * D_ * 2);
  __bf16* vmb   = (__bf16*)alloc((size_t)NW_ * D_ * 2);
  float*  simf  = (float*)alloc((size_t)NW_ * NW_ * 4);
  int*    idxb  = (int*)alloc((size_t)NW_ * TK_ * 4);
  __bf16* msgb  = (__bf16*)alloc(LD2);
  float*  mrgf  = (float*)alloc((size_t)L_ * D_ * 4);
  __bf16* catb  = (__bf16*)alloc((size_t)L_ * D2_ * 2);
  float*  tchw  = (float*)alloc((size_t)L_ * D2_ * 4);
  __bf16* t2b   = (__bf16*)alloc((size_t)L_ * D2_ * 2);
  float*  tmpf  = (float*)alloc((size_t)L_ * D_ * 4);

  // 1) casts
  cast_xs_kernel<<<(L_ * D_ + 255) / 256, 256, 0, stream>>>(x, src, xfb, sfb, catb);
  cast_w_kernel<<<(D_ * D_ + 255) / 256, 256, 0, stream>>>(Wq, wqb, D_ * D_);
  cast_w_kernel<<<(D_ * D_ + 255) / 256, 256, 0, stream>>>(Wk, wkb, D_ * D_);
  cast_w_kernel<<<(D_ * D_ + 255) / 256, 256, 0, stream>>>(Wv, wvb, D_ * D_);
  cast_w_kernel<<<(D_ * D_ + 255) / 256, 256, 0, stream>>>(Wm, wmb_, D_ * D_);
  cast_w_kernel<<<(D2_ * D2_ + 255) / 256, 256, 0, stream>>>(fc1w, f1wb, D2_ * D2_);
  cast_w_kernel<<<(D_ * D2_ + 255) / 256, 256, 0, stream>>>(fc2w, f2wb, D_ * D2_);

  // 2) QKV projections (windowed epilogue), 64x64 tiles
  dim3 gq(D_ / 64, L_ / 64);
  gemm_wmma_kernel<EP_WIN_BF16><<<gq, 32, 0, stream>>>(xfb, wqb, nullptr, nullptr, qwin, L_, D_, D_);
  gemm_wmma_kernel<EP_WIN_BF16><<<gq, 32, 0, stream>>>(sfb, wkb, nullptr, nullptr, kwin, L_, D_, D_);
  gemm_wmma_kernel<EP_WIN_BF16><<<gq, 32, 0, stream>>>(sfb, wvb, nullptr, nullptr, vwin, L_, D_, D_);

  // 3) window means, sim, top-k
  means_kernel<<<NW_, 256, 0, stream>>>(qwin, kwin, vwin, qmb, kmb, vmb);
  gemm_wmma_kernel<EP_F32><<<dim3(NW_ / 64, NW_ / 64), 32, 0, stream>>>(
      qmb, kmb, nullptr, simf, nullptr, NW_, NW_, D_);
  topk_kernel<<<NW_, 32, 0, stream>>>(simf, idxb);

  // 4) attention (window x head)
  attn_kernel<<<dim3(NW_, NH_), 32, 0, stream>>>(qwin, kwin, vwin, kmb, vmb, idxb, msgb);

  // 5) merge projection + LN1 -> concat second half
  gemm_wmma_kernel<EP_F32><<<gq, 32, 0, stream>>>(msgb, wmb_, nullptr, mrgf, nullptr, L_, D_, D_);
  ln1_kernel<<<L_, 256, 0, stream>>>(mrgf, n1g, n1b, catb);

  // 6) MLP: fc1 (+bias,relu, NCHW out) -> dwconv+gelu -> fc2 (+bias)
  gemm_wmma_kernel<EP_BIAS_RELU_T><<<dim3(D2_ / 64, L_ / 64), 32, 0, stream>>>(
      catb, f1wb, fc1b, tchw, nullptr, L_, D2_, D2_);
  dwconv_kernel<<<dim3(7, 7, D2_), dim3(16, 16), 0, stream>>>(tchw, dww, dwb, t2b);
  gemm_wmma_kernel<EP_BIAS_F32><<<dim3(D_ / 64, L_ / 64), 32, 0, stream>>>(
      t2b, f2wb, fc2b, tmpf, nullptr, L_, D_, D2_);

  // 7) LN2 + residual + transpose back to NCHW
  ln2_kernel<<<L_, 256, 0, stream>>>(tmpf, x, n2g, n2b, out);
}
